// SpikingRecurrent_49125835931932
// MI455X (gfx1250) — matrix operand order repaired and checked by reading, not compiled, not run on hardware
//
#include <hip/hip_runtime.h>

// ---------------------------------------------------------------------------
// SpikingRecurrent for MI455X (gfx1250, wave32, WMMA bf16)
//   Phase 0: zero split-K accumulator (T*B x HIDDEN f32) in d_ws
//   Phase 1: encoder GEMM  ENC += X_p @ We_p.T   via v_wmma_f32_16x16x32_bf16
//            split-K over (p, K-chunk); 128-thread blocks (4 waves), each wave
//            2 M-tiles (timesteps) x 4 N-tiles -> 3 b128 + 6 cvt per WMMA;
//            unguarded main loop + peeled guarded tail; atomic f32 reduction
//   Phase 2: sequential LIF scan (1 workgroup), recurrent matmuls via WMMA
// ---------------------------------------------------------------------------

typedef __attribute__((ext_vector_type(16))) __bf16 v16bf;
typedef __attribute__((ext_vector_type(8)))  __bf16 v8bf;
typedef __attribute__((ext_vector_type(8)))  float  v8f;

#define HW        89960      // 260*346
#define KTOT      179920     // 2*HW
#define T_STEPS   20
#define BATCH     16
#define HIDDEN    256
#define NCLS      10
#define KCHUNK    8192
#define NCHUNK_P  11         // ceil(HW / KCHUNK); last chunk has 8040 valid (mult of 8)

__device__ __forceinline__ v16bf pack16(const float4& f0, const float4& f1,
                                        const float4& f2, const float4& f3) {
  v16bf r;
  r[0]  = (__bf16)f0.x; r[1]  = (__bf16)f0.y; r[2]  = (__bf16)f0.z; r[3]  = (__bf16)f0.w;
  r[4]  = (__bf16)f1.x; r[5]  = (__bf16)f1.y; r[6]  = (__bf16)f1.z; r[7]  = (__bf16)f1.w;
  r[8]  = (__bf16)f2.x; r[9]  = (__bf16)f2.y; r[10] = (__bf16)f2.z; r[11] = (__bf16)f2.w;
  r[12] = (__bf16)f3.x; r[13] = (__bf16)f3.y; r[14] = (__bf16)f3.z; r[15] = (__bf16)f3.w;
  return r;
}

// A fragment (16x32 MxK bf16): xa already offset by lane-half (+0/+8).
// Lane supplies K runs {kk..kk+7} and {kk+16..kk+23} relative to xa.
__device__ __forceinline__ v16bf load_afrag(const float* __restrict__ xa, int kk) {
  const float4* x4 = (const float4*)(xa + kk);
  return pack16(x4[0], x4[1], x4[4], x4[5]);
}

// B fragment (32x16 KxN bf16): w already offset by lane-half (+0/+8); K runs {kk..kk+7},{kk+16..kk+23}.
__device__ __forceinline__ v16bf load_bfrag(const float* __restrict__ w, int kk) {
  const float4* w4 = (const float4*)(w + kk);
  return pack16(w4[0], w4[1], w4[4], w4[5]);
}

// Guarded variants for the single peeled tail step (8-run granularity, vlim mult of 8).
__device__ __forceinline__ v16bf load_afrag_g(const float* __restrict__ xa, int kk,
                                              int rbase, int vlim) {
  const float4 z4 = make_float4(0.f, 0.f, 0.f, 0.f);
  float4 a0 = z4, a1 = z4, a2 = z4, a3 = z4;
  if (rbase < vlim)      { a0 = *(const float4*)(xa + kk);      a1 = *(const float4*)(xa + kk + 4);  }
  if (rbase + 16 < vlim) { a2 = *(const float4*)(xa + kk + 16); a3 = *(const float4*)(xa + kk + 20); }
  return pack16(a0, a1, a2, a3);
}
__device__ __forceinline__ v16bf load_bfrag_g(const float* __restrict__ w, int kk,
                                              int rbase, int vlim) {
  const float4 z4 = make_float4(0.f, 0.f, 0.f, 0.f);
  float4 f0 = z4, f1 = z4, f2 = z4, f3 = z4;
  if (rbase < vlim)      { f0 = *(const float4*)(w + kk);      f1 = *(const float4*)(w + kk + 4);  }
  if (rbase + 16 < vlim) { f2 = *(const float4*)(w + kk + 16); f3 = *(const float4*)(w + kk + 20); }
  return pack16(f0, f1, f2, f3);
}

__global__ void zero_f32_kernel(float* __restrict__ p, int n) {
  int i = blockIdx.x * blockDim.x + threadIdx.x;
  if (i < n) p[i] = 0.f;
}

#define WMMA_BF16(A, B, C) \
  __builtin_amdgcn_wmma_f32_16x16x32_bf16(false, (A), false, (B), (short)0, (C), false, false)

// ---------------------------------------------------------------------------
// Phase 1: grid = (2*NCHUNK_P, T_STEPS/2), block = 128 (4 waves).
// Wave wv: M-tiles t0,t1 x N-tiles {64wv+0, +16, +32, +48}.
// ---------------------------------------------------------------------------
__global__ __launch_bounds__(128) void enc_gemm_kernel(const float* __restrict__ voxel,
                                                       const float* __restrict__ We,
                                                       float* __restrict__ encacc) {
  const int lane = threadIdx.x & 31;
  const int wv   = threadIdx.x >> 5;              // 0..3
  const int hiH  = lane >> 4;                     // lane half
  const int t0   = blockIdx.y * 2;                // two M-tiles per block
  const int t1   = t0 + 1;
  const int p    = blockIdx.x / NCHUNK_P;         // polarity half of K
  const int kc   = (blockIdx.x % NCHUNK_P) * KCHUNK;
  const int vlim = min(KCHUNK, HW - kc);          // 8192 or 8040
  const int kmain = vlim & ~31;                   // unguarded trip count
  const int aoff = hiH ? 8 : 0;
  const int boff = hiH ? 8 : 0;

  const int b = lane & 15;                        // A: M index = batch
  const size_t xbase = (size_t)(b * 2 + p) * T_STEPS * HW + (size_t)kc;
  const float* __restrict__ xa0 = voxel + xbase + (size_t)t0 * HW + aoff;
  const float* __restrict__ xa1 = voxel + xbase + (size_t)t1 * HW + aoff;

  const int nbase = wv * 64;
  const int nA = nbase + (lane & 15);             // four N-tiles: N = lane%16 + 16*i
  const size_t wstep = (size_t)16 * KTOT;
  const float* __restrict__ w0 = We + (size_t)nA * KTOT + (size_t)p * HW + kc + boff;
  const float* __restrict__ w1 = w0 + wstep;
  const float* __restrict__ w2 = w1 + wstep;
  const float* __restrict__ w3 = w2 + wstep;

  v8f acc00 = {}, acc01 = {}, acc02 = {}, acc03 = {};   // t0 x N-tiles 0..3
  v8f acc10 = {}, acc11 = {}, acc12 = {}, acc13 = {};   // t1 x N-tiles 0..3

  #pragma unroll 2
  for (int kk = 0; kk < kmain; kk += 32) {
    __builtin_prefetch(w0 + kk + 256, 0, 1);      // speculative; OOB is dropped
    __builtin_prefetch(w1 + kk + 256, 0, 1);
    __builtin_prefetch(w2 + kk + 256, 0, 1);
    __builtin_prefetch(w3 + kk + 256, 0, 1);
    const v16bf af0 = load_afrag(xa0, kk);
    const v16bf af1 = load_afrag(xa1, kk);
    const v16bf b0  = load_bfrag(w0, kk);
    const v16bf b1  = load_bfrag(w1, kk);
    const v16bf b2  = load_bfrag(w2, kk);
    const v16bf b3  = load_bfrag(w3, kk);
    acc00 = WMMA_BF16(af0, b0, acc00);
    acc01 = WMMA_BF16(af0, b1, acc01);
    acc02 = WMMA_BF16(af0, b2, acc02);
    acc03 = WMMA_BF16(af0, b3, acc03);
    acc10 = WMMA_BF16(af1, b0, acc10);
    acc11 = WMMA_BF16(af1, b1, acc11);
    acc12 = WMMA_BF16(af1, b2, acc12);
    acc13 = WMMA_BF16(af1, b3, acc13);
  }

  if (vlim & 31) {                                // single guarded tail step
    const int kk = kmain;
    const v16bf af0 = load_afrag_g(xa0, kk, kk + aoff, vlim);
    const v16bf af1 = load_afrag_g(xa1, kk, kk + aoff, vlim);
    const v16bf b0  = load_bfrag_g(w0, kk, kk + boff, vlim);
    const v16bf b1  = load_bfrag_g(w1, kk, kk + boff, vlim);
    const v16bf b2  = load_bfrag_g(w2, kk, kk + boff, vlim);
    const v16bf b3  = load_bfrag_g(w3, kk, kk + boff, vlim);
    acc00 = WMMA_BF16(af0, b0, acc00);
    acc01 = WMMA_BF16(af0, b1, acc01);
    acc02 = WMMA_BF16(af0, b2, acc02);
    acc03 = WMMA_BF16(af0, b3, acc03);
    acc10 = WMMA_BF16(af1, b0, acc10);
    acc11 = WMMA_BF16(af1, b1, acc11);
    acc12 = WMMA_BF16(af1, b2, acc12);
    acc13 = WMMA_BF16(af1, b3, acc13);
  }

  // C/D layout: VGPR j -> M = j + 8*hiH, N = lane%16 (+16 per tile). Split-K atomics.
  #pragma unroll
  for (int j = 0; j < 8; ++j) {
    const int mm = (hiH ? 8 : 0) + j;
    float* r0 = encacc + ((size_t)t0 * BATCH + mm) * HIDDEN + nA;
    float* r1 = encacc + ((size_t)t1 * BATCH + mm) * HIDDEN + nA;
    unsafeAtomicAdd(r0,      acc00[j]);
    unsafeAtomicAdd(r0 + 16, acc01[j]);
    unsafeAtomicAdd(r0 + 32, acc02[j]);
    unsafeAtomicAdd(r0 + 48, acc03[j]);
    unsafeAtomicAdd(r1,      acc10[j]);
    unsafeAtomicAdd(r1 + 16, acc11[j]);
    unsafeAtomicAdd(r1 + 32, acc12[j]);
    unsafeAtomicAdd(r1 + 48, acc13[j]);
  }
}

// ---------------------------------------------------------------------------
// Phase 2: sequential LIF scan, one workgroup (8 waves), state in LDS.
//   v' = 0.9 v + 0.1 i ; i' = 0.8 i ; z_new = (v' > 1) ; v = (1-z_new) v'
//   i = i' + enc@Win.T + z_old@Wrec.T ; out = z_new@Wd.T + bd
// ---------------------------------------------------------------------------
__global__ __launch_bounds__(256) void lif_scan_kernel(const float* __restrict__ encacc,
                                                       const float* __restrict__ be,
                                                       const float* __restrict__ Win,
                                                       const float* __restrict__ Wrec,
                                                       const float* __restrict__ Wd,
                                                       const float* __restrict__ bd,
                                                       float* __restrict__ out) {
  __shared__ float  v_s[BATCH * HIDDEN];
  __shared__ float  i_s[BATCH * HIDDEN];
  __shared__ __bf16 z_bf[BATCH * HIDDEN];   // old spikes (WMMA A operand)
  __shared__ __bf16 e_bf[BATCH * HIDDEN];   // encoded input (WMMA A operand)

  const int tid  = threadIdx.x;
  const int lane = tid & 31;
  const int wv   = tid >> 5;
  const int hiH  = lane >> 4;
  const int n0   = wv * 32;
  const int nA   = n0 + (lane & 15);
  const int nB   = nA + 16;
  const int bA   = lane & 15;

  for (int idx = tid; idx < BATCH * HIDDEN; idx += 256) {
    v_s[idx] = 0.f; i_s[idx] = 0.f; z_bf[idx] = (__bf16)0.f;
  }
  __syncthreads();

  for (int t = 0; t < T_STEPS; ++t) {
    // S1: enc = relu(acc + be)
    for (int idx = tid; idx < BATCH * HIDDEN; idx += 256) {
      float e = encacc[(size_t)t * BATCH * HIDDEN + idx] + be[idx & (HIDDEN - 1)];
      e_bf[idx] = (__bf16)(e > 0.f ? e : 0.f);
    }
    __syncthreads();

    // S2: r = enc@Win.T + z_old@Wrec.T (each wave: two 16x16 N-tiles, K=256)
    v8f acc0 = {}; v8f acc1 = {};
    #pragma unroll
    for (int kk = 0; kk < HIDDEN; kk += 32) {
      const int k0 = kk + (hiH ? 8 : 0);
      const __bf16* __restrict__ er = e_bf + bA * HIDDEN + k0;
      const __bf16* __restrict__ zr = z_bf + bA * HIDDEN + k0;
      const v8bf elo = *(const v8bf*)(er);
      const v8bf ehi = *(const v8bf*)(er + 16);
      const v8bf zlo = *(const v8bf*)(zr);
      const v8bf zhi = *(const v8bf*)(zr + 16);
      const v16bf aE = __builtin_shufflevector(elo, ehi, 0,1,2,3,4,5,6,7,8,9,10,11,12,13,14,15);
      const v16bf aZ = __builtin_shufflevector(zlo, zhi, 0,1,2,3,4,5,6,7,8,9,10,11,12,13,14,15);
      const int kb = kk + (hiH ? 8 : 0);
      const v16bf bWinA = load_bfrag(Win  + (size_t)nA * HIDDEN, kb);
      const v16bf bWinB = load_bfrag(Win  + (size_t)nB * HIDDEN, kb);
      const v16bf bWrcA = load_bfrag(Wrec + (size_t)nA * HIDDEN, kb);
      const v16bf bWrcB = load_bfrag(Wrec + (size_t)nB * HIDDEN, kb);
      acc0 = WMMA_BF16(aE, bWinA, acc0);
      acc0 = WMMA_BF16(aZ, bWrcA, acc0);
      acc1 = WMMA_BF16(aE, bWinB, acc1);
      acc1 = WMMA_BF16(aZ, bWrcB, acc1);
    }
    __syncthreads();   // all z_bf reads (A fragments) complete before state update

    // S3: LIF update; each lane exclusively owns (m, nA) and (m, nB)
    #pragma unroll
    for (int j = 0; j < 8; ++j) {
      const int mm = (hiH ? 8 : 0) + j;
      {
        const int idx = mm * HIDDEN + nA;
        const float vo = v_s[idx], io = i_s[idx];
        const float vd = 0.9f * vo + 0.1f * io;
        const float zf = (vd > 1.0f) ? 1.f : 0.f;
        v_s[idx]  = (1.f - zf) * vd;
        i_s[idx]  = 0.8f * io + acc0[j];
        z_bf[idx] = (__bf16)zf;
      }
      {
        const int idx = mm * HIDDEN + nB;
        const float vo = v_s[idx], io = i_s[idx];
        const float vd = 0.9f * vo + 0.1f * io;
        const float zf = (vd > 1.0f) ? 1.f : 0.f;
        v_s[idx]  = (1.f - zf) * vd;
        i_s[idx]  = 0.8f * io + acc1[j];
        z_bf[idx] = (__bf16)zf;
      }
    }
    __syncthreads();

    // S4: decoder out[t,b,c] = z_new[b,:] . Wd[c,:] + bd[c]
    if (tid < BATCH * NCLS) {
      const int b = tid / NCLS, c = tid % NCLS;
      float s = bd[c];
      const __bf16* __restrict__ zr = z_bf + b * HIDDEN;
      #pragma unroll 8
      for (int h = 0; h < HIDDEN; ++h) s += (float)zr[h] * Wd[c * HIDDEN + h];
      out[((size_t)t * BATCH + b) * NCLS + c] = s;
    }
    __syncthreads();
  }
}

// ---------------------------------------------------------------------------
extern "C" void kernel_launch(void* const* d_in, const int* in_sizes, int n_in,
                              void* d_out, int out_size, void* d_ws, size_t ws_size,
                              hipStream_t stream) {
  const float* voxel = (const float*)d_in[0];
  const float* We    = (const float*)d_in[1];
  const float* be    = (const float*)d_in[2];
  const float* Win   = (const float*)d_in[3];
  const float* Wrec  = (const float*)d_in[4];
  const float* Wd    = (const float*)d_in[5];
  const float* bd    = (const float*)d_in[6];
  float* out    = (float*)d_out;
  float* encacc = (float*)d_ws;                   // T*B*HIDDEN f32 = 320 KiB

  const int nacc = T_STEPS * BATCH * HIDDEN;      // 81920
  zero_f32_kernel<<<(nacc + 255) / 256, 256, 0, stream>>>(encacc, nacc);

  dim3 grid(2 * NCHUNK_P, T_STEPS / 2);           // 22 K-chunks x 10 M-pairs = 220 WGs
  enc_gemm_kernel<<<grid, 128, 0, stream>>>(voxel, We, encacc);

  lif_scan_kernel<<<1, 256, 0, stream>>>(encacc, be, Win, Wrec, Wd, bd, out);
}
